// HMMTrajNet_10660108829204
// MI455X (gfx1250) — matrix operand order repaired and checked by reading, not compiled, not run on hardware
//
#include <hip/hip_runtime.h>
#include <hip/hip_bf16.h>
#include <math.h>

typedef __attribute__((ext_vector_type(16))) __bf16 v16bf;
typedef __attribute__((ext_vector_type(8)))  float  v8f;

#define BATCH   256
#define T1      513
#define MAXT    512
#define SDIM    64
#define NCOLS   304              // 256 action + 32 stop + 16 start
#define NTILES  (NCOLS / 16)     // 19
#define TOKENS  (BATCH * T1)     // 131328
#define MTILES  (TOKENS / 16)    // 8208
#define LSTRIDE 308              // 304 + 4 pad (LDS bank-conflict pad)
#define WFRAG_ELEMS (NTILES * 2 * 32 * 16)   // 19456 bf16 = 38912 bytes

// ---------------------------------------------------------------- helpers ---
__device__ __forceinline__ unsigned short f2bf_rne(float f) {
    union { float f; unsigned u; } v; v.f = f;
    unsigned r = v.u + 0x7FFFu + ((v.u >> 16) & 1u);   // round-to-nearest-even
    return (unsigned short)(r >> 16);
}

// logsumexp across 16 lanes (lane = state index), lanes 0..15 active
__device__ __forceinline__ float lse16(float v) {
    float m = v;
    #pragma unroll
    for (int o = 8; o >= 1; o >>= 1) m = fmaxf(m, __shfl_xor(m, o, 16));
    float s = __expf(v - m);
    #pragma unroll
    for (int o = 8; o >= 1; o >>= 1) s += __shfl_xor(s, o, 16);
    return m + __logf(s);
}

// ------------------------------------------- kernel 0: pack W to fragments ---
// Emits W in per-lane WMMA B-fragment order:
//   Wfrag[nt][ks][lane][e]  (e = 0..15 bf16, one full B fragment per lane)
// B order used here (16x16x32 bf16, contiguous k, not the ISA table order): lane n = L&15, half = L>>4 selects K 0-15 / 16-31,
// VGPR j packs K = 2j+16*half and 2j+1+16*half.
__global__ void pack_w_kernel(const float* __restrict__ Wa, const float* __restrict__ ba,
                              const float* __restrict__ Ws, const float* __restrict__ bs,
                              const float* __restrict__ Wt, const float* __restrict__ bt,
                              unsigned short* __restrict__ Wfrag, float* __restrict__ biasp) {
    int idx = blockIdx.x * blockDim.x + threadIdx.x;
    int nthr = gridDim.x * blockDim.x;
    for (int i = idx; i < WFRAG_ELEMS; i += nthr) {
        const int e    = i & 15;
        const int L    = (i >> 4) & 31;
        const int ks   = (i >> 9) & 1;
        const int nt   = i >> 10;
        const int n    = L & 15;
        const int half = L >> 4;
        const int j    = e >> 1;
        const int lo   = e & 1;
        const int k    = ks * 32 + 2 * j + 16 * half + lo;   // 0..63
        const int col  = nt * 16 + n;                        // 0..303
        float v = (col < 256) ? Wa[k * 256 + col]
                : (col < 288) ? Ws[k * 32 + (col - 256)]
                              : Wt[k * 16 + (col - 288)];
        Wfrag[i] = f2bf_rne(v);
    }
    for (int c = idx; c < NCOLS; c += nthr) {
        biasp[c] = (c < 256) ? ba[c] : (c < 288) ? bs[c - 256] : bt[c - 288];
    }
}

// ------------------------------------- kernel 1: fused WMMA GEMM + logsoftmax
// one wave (32 lanes) per 16-token tile
__global__ __launch_bounds__(32) void logits_kernel(
    const float* __restrict__ s_i, const int* __restrict__ actions,
    const unsigned short* __restrict__ WfragU, const float* __restrict__ biasp,
    float* __restrict__ aSel, float* __restrict__ stopS,
    float* __restrict__ stopC, float* __restrict__ startL)
{
    __shared__ float sLog[16][LSTRIDE];
    const int lane = threadIdx.x;
    const int m    = lane & 15;
    const int half = lane >> 4;
    const int tokBase = blockIdx.x * 16;
    const v16bf* Wf = (const v16bf*)WfragU;   // 32-byte fragments, coalesced

    // --- A fragments (token rows), 16-bit A 16x32, contiguous-k order used here:
    // VGPR j: K = 2j + 16*half (contiguous, matches packed B)
    v16bf a0, a1;
    {
        const float* Arow = s_i + (size_t)(tokBase + m) * SDIM;
        #pragma unroll
        for (int j = 0; j < 8; ++j) {
            const int k = (j * 2) + half * 16;
            float2 p0 = *(const float2*)(Arow + k);        // K-step 0 (k in 0..31)
            float2 p1 = *(const float2*)(Arow + 32 + k);   // K-step 1 (k in 32..63)
            a0[2 * j] = (__bf16)p0.x;  a0[2 * j + 1] = (__bf16)p0.y;
            a1[2 * j] = (__bf16)p1.x;  a1[2 * j + 1] = (__bf16)p1.y;
        }
    }

    // --- GEMM over 19 column tiles; B fragments pre-packed: 1 vector load each
    const int n = m;
    for (int nt = 0; nt < NTILES; ++nt) {
        const int c0 = nt * 16;
        const v16bf b0 = Wf[(nt * 2 + 0) * 32 + lane];
        const v16bf b1 = Wf[(nt * 2 + 1) * 32 + lane];
        v8f acc = {};
        acc = __builtin_amdgcn_wmma_f32_16x16x32_bf16(false, a0, false, b0,
                                                      (short)0, acc, false, false);
        acc = __builtin_amdgcn_wmma_f32_16x16x32_bf16(false, a1, false, b1,
                                                      (short)0, acc, false, false);
        const float bias = biasp[c0 + n];
        // C/D layout: VGPR r -> row m = r + 8*half, column n = lane&15
        #pragma unroll
        for (int r = 0; r < 8; ++r)
            sLog[r + 8 * half][c0 + n] = acc[r] + bias;
    }
    __syncthreads();

    // --- fused grouped log-softmaxes; lane handles token m, groups half*8..+7
    const int token = tokBase + m;
    const int b = token / T1;
    const int t = token - b * T1;
    const int act = (t < MAXT) ? actions[(size_t)b * MAXT + t] : 0;
    const size_t obase = (size_t)token * 16;

    #pragma unroll
    for (int gi = 0; gi < 8; ++gi) {
        const int g = half * 8 + gi;
        // action group g: 16 logits
        const float* row = &sLog[m][g * 16];
        float mx = row[0];
        #pragma unroll
        for (int i = 1; i < 16; ++i) mx = fmaxf(mx, row[i]);
        float s = 0.f;
        #pragma unroll
        for (int i = 0; i < 16; ++i) s += __expf(row[i] - mx);
        aSel[obase + g] = row[act] - mx - __logf(s);
        // stop pair g: [stop, continue]
        float x0 = sLog[m][256 + g * 2];
        float x1 = sLog[m][256 + g * 2 + 1];
        float mm = fmaxf(x0, x1);
        float l  = mm + __logf(__expf(x0 - mm) + __expf(x1 - mm));
        stopS[obase + g] = x0 - l;
        stopC[obase + g] = x1 - l;
    }
    // start log-softmax over 16 (computed redundantly per half, writes split)
    {
        float mx = sLog[m][288];
        #pragma unroll
        for (int i = 1; i < 16; ++i) mx = fmaxf(mx, sLog[m][288 + i]);
        float s = 0.f;
        #pragma unroll
        for (int i = 0; i < 16; ++i) s += __expf(sLog[m][288 + i] - mx);
        const float lse = mx + __logf(s);
        #pragma unroll
        for (int gi = 0; gi < 8; ++gi) {
            const int g = half * 8 + gi;
            startL[obase + g] = sLog[m][288 + g] - lse;
        }
    }
}

// ------------------------------------------- kernel 2: HMM forward scan ----
// one block per batch element; lanes 0..15 = the 16 HMM states
__global__ __launch_bounds__(32) void hmm_scan_kernel(
    const float* __restrict__ aSel, const float* __restrict__ stopS,
    const float* __restrict__ stopC, const float* __restrict__ startL,
    const int* __restrict__ lengths, float* __restrict__ partial)
{
    const int b = blockIdx.x;
    const int lane = threadIdx.x;
    if (lane >= 16) return;
    const int nb = lane;
    const int len = lengths[b];
    const size_t base = (size_t)b * T1 * 16;

    // f0 = start_logps[:,0] + a_sel[:,0]
    float f = startL[base + nb] + aSel[base + nb];
    for (int t = 1; t <= len - 1; ++t) {
        const size_t off = base + (size_t)t * 16 + nb;
        const float beta = stopS[off];
        const float cont = stopC[off];
        const float st   = startL[off];
        const float at   = aSel[off];
        const float lse  = lse16(f + beta);       // logsumexp over states
        const float u = f + cont;                 // stay in option
        const float v = lse + st;                 // restart option
        const float mm = fmaxf(u, v);
        f = mm + __logf(__expf(u - mm) + __expf(v - mm)) + at;
    }
    // x0 = fs[len-1] (== f) ; x1 = stop_logps[:, len, :, STOP]
    const float x1 = stopS[base + (size_t)len * 16 + nb];
    const float r = lse16(f + x1);
    if (lane == 0) partial[b] = r;
}

// --------------------------------------------- kernel 3: final reduction ---
__global__ __launch_bounds__(256) void reduce_kernel(const float* __restrict__ partial,
                                                     float* __restrict__ out) {
    __shared__ float s[BATCH];
    const int i = threadIdx.x;
    s[i] = partial[i];
    __syncthreads();
    for (int st = BATCH / 2; st > 0; st >>= 1) {
        if (i < st) s[i] += s[i + st];
        __syncthreads();
    }
    if (i == 0) out[0] = -s[0];
}

// ---------------------------------------------------------------- launch ---
extern "C" void kernel_launch(void* const* d_in, const int* in_sizes, int n_in,
                              void* d_out, int out_size, void* d_ws, size_t ws_size,
                              hipStream_t stream) {
    (void)in_sizes; (void)n_in; (void)out_size; (void)ws_size;
    const float* s_i     = (const float*)d_in[0];
    const int*   actions = (const int*)  d_in[1];
    const int*   lengths = (const int*)  d_in[2];
    const float* W_action = (const float*)d_in[3];
    const float* b_action = (const float*)d_in[4];
    const float* W_stop   = (const float*)d_in[5];
    const float* b_stop   = (const float*)d_in[6];
    const float* W_start  = (const float*)d_in[7];
    const float* b_start  = (const float*)d_in[8];

    char* ws = (char*)d_ws;
    unsigned short* Wfrag = (unsigned short*)ws;           // 38912 B (frag order)
    float* biasp = (float*)(ws + 39168);                   // 304*4 = 1216 B
    float* aSel  = (float*)(ws + 40448);                   // per-token log-probs
    const size_t NTOK16 = (size_t)TOKENS * 16;             // 2,101,248 floats each
    float* stopS  = aSel  + NTOK16;
    float* stopC  = stopS + NTOK16;
    float* startL = stopC + NTOK16;
    float* partial = startL + NTOK16;                      // 256 floats

    pack_w_kernel<<<76, 256, 0, stream>>>(W_action, b_action, W_stop, b_stop,
                                          W_start, b_start, Wfrag, biasp);
    logits_kernel<<<MTILES, 32, 0, stream>>>(s_i, actions, Wfrag, biasp,
                                             aSel, stopS, stopC, startL);
    hmm_scan_kernel<<<BATCH, 32, 0, stream>>>(aSel, stopS, stopC, startL,
                                              lengths, partial);
    reduce_kernel<<<1, 256, 0, stream>>>(partial, (float*)d_out);
}
